// GATv2_14491219656879
// MI455X (gfx1250) — matrix operand order repaired and checked by reading, not compiled, not run on hardware
//
#include <hip/hip_runtime.h>
#include <hip/hip_bf16.h>
#include <math.h>

// ---------------------------------------------------------------------------
// GATv2 (3 layers) for gfx1250 / MI455X.
//  - Dense transforms: v_wmma_f32_16x16x32_f16, LDS-staged 64x64 block tiles,
//    2 WMMA tiles per wave sharing one A fragment. Loaders are phase-split
//    (batched global loads -> f16 pack -> ds_store_b128).
//  - Edge softmax/aggregation: 3 streaming float4 passes, u32 atomic-max
//    (monotonic float encoding) + global_atomic_add_f32.
// Assumes K even (58 / 256 / 256 here).
// ---------------------------------------------------------------------------

typedef _Float16 v16h __attribute__((ext_vector_type(16)));
typedef _Float16 h8   __attribute__((ext_vector_type(8)));
typedef float    v8f  __attribute__((ext_vector_type(8)));

#define NEG_SLOPE 0.2f

__device__ __forceinline__ unsigned f2ord(float f) {
  unsigned u = __float_as_uint(f);
  return (u & 0x80000000u) ? ~u : (u | 0x80000000u);
}
__device__ __forceinline__ float ord2f(unsigned e) {
  return (e & 0x80000000u) ? __uint_as_float(e & 0x7fffffffu)
                           : __uint_as_float(~e);
}
__device__ __forceinline__ int imin(int a, int b) { return a < b ? a : b; }

// ---------------------------------------------------------------------------
// Fills
// ---------------------------------------------------------------------------
__global__ void __launch_bounds__(256) fill_f32(float* __restrict__ p, float v, long n) {
  long i = (long)blockIdx.x * blockDim.x + threadIdx.x;
  if (i < n) p[i] = v;
}
__global__ void __launch_bounds__(256) fill_u32(unsigned* __restrict__ p, unsigned v, long n) {
  long i = (long)blockIdx.x * blockDim.x + threadIdx.x;
  if (i < n) p[i] = v;
}

// ---------------------------------------------------------------------------
// GEMM: Y[N,M] = X[N,K] @ W[K,M] + bias[M]   (fp32 -> f16 WMMA -> fp32)
//
// Block = 256 threads = 8 waves, 64(row) x 64(col) tile.
// Wave w: rt = w>>1 (row tile), cp = w&1 (32-col pair) -> two 16x16 tiles
// sharing one A fragment; 2 x v_wmma_f32_16x16x32_f16 per K-step per wave.
//
// LDS per K-step (32):
//   lA [64 rows][32 k]  f16   - A loader: thread t -> row=t>>2, k0=8*(t&3);
//                               4 x float2 global loads (clamped addr,
//                               value-masked), pack 8 halves, ds_store_b128.
//   lBt[64 cols][32 k]  f16   - B loader: thread t -> col=t&63, k0=8*(t>>6);
//                               8 coalesced b32 loads (consecutive cols per j),
//                               pack 8 halves, ds_store_b128.
// Fragments (contiguous -> ds_load_b128):
//   A: lane(m=l&15,h=l>>4): halves {16h..16h+15} of row m
//   B: lane n=l&15: halves {16h..16h+15} of column n (transposed tile)
// C/D: lane l -> col n=l&15 ; vgpr r -> row m = r + 8*(l>>4).
// ---------------------------------------------------------------------------
__global__ void __launch_bounds__(256) gemm_bias_wmma(
    const float* __restrict__ X, const float* __restrict__ W,
    const float* __restrict__ bias, float* __restrict__ Y,
    int N, int K, int M, int colBlocks)
{
  __shared__ _Float16 lA[64][32];
  __shared__ _Float16 lBt[64][32];

  const int bidRow = blockIdx.x / colBlocks;
  const int bidCol = blockIdx.x % colBlocks;
  const int rowB = bidRow * 64;
  const int colB = bidCol * 64;

  const int t    = threadIdx.x;
  const int wave = t >> 5;
  const int lane = t & 31;
  const int rt = wave >> 1;            // row tile 0..3
  const int cp = wave & 1;             // col pair 0..1
  const int m  = lane & 15;
  const int hh = lane >> 4;

  // A loader coords: one 8-k chunk of one row per thread
  const int arow   = t >> 2;           // 0..63
  const int ak0    = (t & 3) * 8;      // 0,8,16,24
  const int growA  = rowB + arow;
  const int growAc = imin(growA, N - 1);
  // B loader coords: one 8-k chunk of one column per thread
  const int bcolL  = t & 63;           // 0..63
  const int bk0    = (t >> 6) * 8;     // 0,8,16,24
  const int gcolB  = colB + bcolL;
  const int gcolBc = imin(gcolB, M - 1);

  v8f acc0 = {}, acc1 = {};
  for (int kb = 0; kb < K; kb += 32) {
    __syncthreads();                   // previous compute done before overwrite

    // ---- A loader: batched float2 loads, then pack + one ds_store_b128 ----
    if (growA < N && kb + 32 < K)
      __builtin_prefetch(&X[(long)growA * K + kb + 32 + ak0], 0, 3);
    {
      float2 va[4];
#pragma unroll
      for (int j = 0; j < 4; ++j) {
        const int gk = kb + ak0 + 2 * j;
        va[j] = *(const float2*)&X[(long)growAc * K + imin(gk, K - 2)];
      }
      h8 ah;
#pragma unroll
      for (int j = 0; j < 4; ++j) {
        const int gk = kb + ak0 + 2 * j;
        const bool ok0 = (growA < N) & (gk < K);
        const bool ok1 = (growA < N) & (gk + 1 < K);
        ah[2 * j]     = (_Float16)(ok0 ? va[j].x : 0.0f);
        ah[2 * j + 1] = (_Float16)(ok1 ? va[j].y : 0.0f);
      }
      *(h8*)&lA[arow][ak0] = ah;
    }
    // ---- B loader: 8 coalesced b32 loads, then pack + one ds_store_b128 ----
    {
      float vb[8];
#pragma unroll
      for (int j = 0; j < 8; ++j) {
        const int gk = kb + bk0 + j;
        vb[j] = W[(long)imin(gk, K - 1) * M + gcolBc];
      }
      h8 bh;
#pragma unroll
      for (int j = 0; j < 8; ++j) {
        const int gk = kb + bk0 + j;
        const bool ok = (gk < K) & (gcolB < M);
        bh[j] = (_Float16)(ok ? vb[j] : 0.0f);
      }
      *(h8*)&lBt[bcolL][bk0] = bh;
    }
    __syncthreads();

    // ---- fragments from LDS (contiguous -> ds_load_b128) ----
    const h8 a0 = *(const h8*)&lA[16 * rt + m][16 * hh];
    const h8 a1 = *(const h8*)&lA[16 * rt + m][16 * hh + 8];
    const v16h a = __builtin_shufflevector(a0, a1,
        0, 1, 2, 3, 4, 5, 6, 7, 8, 9, 10, 11, 12, 13, 14, 15);
    const v16h b0 = *(const v16h*)&lBt[32 * cp + m][16 * hh];
    const v16h b1 = *(const v16h*)&lBt[32 * cp + 16 + m][16 * hh];

    acc0 = __builtin_amdgcn_wmma_f32_16x16x32_f16(
        false, a, false, b0, (short)0, acc0, false, false);
    acc1 = __builtin_amdgcn_wmma_f32_16x16x32_f16(
        false, a, false, b1, (short)0, acc1, false, false);
  }

  // ---- store D tiles (guarded; uniform WMMA path above is unconditional) ----
  const int col0 = colB + 32 * cp + m;
  const int col1 = col0 + 16;
  const float bv0 = (col0 < M) ? bias[col0] : 0.0f;
  const float bv1 = (col1 < M) ? bias[col1] : 0.0f;
#pragma unroll
  for (int r = 0; r < 8; ++r) {
    const int row = rowB + 16 * rt + r + 8 * hh;
    if (row < N) {
      if (col0 < M) Y[(long)row * M + col0] = acc0[r] + bv0;
      if (col1 < M) Y[(long)row * M + col1] = acc1[r] + bv1;
    }
  }
}

// ---------------------------------------------------------------------------
// Pass A: logits[e,h] = sum_c leakyrelu(xl[src,h,c]+xr[dst,h,c]) * att[h,c]
//         + atomic segment-max into mOrd[dst,h].
// Edges e >= Eorig are implicit self-loops (src = dst = e - Eorig).
// ---------------------------------------------------------------------------
__global__ void __launch_bounds__(256) edge_logits_max(
    const float* __restrict__ xl, const float* __restrict__ xr,
    const int* __restrict__ src, const int* __restrict__ dst,
    const float* __restrict__ att, float* __restrict__ logits,
    unsigned* __restrict__ mOrd, long Eorig, long Etot, int H, int C)
{
  long tid = (long)blockIdx.x * blockDim.x + threadIdx.x;
  if (tid >= Etot * H) return;
  const long e = tid / H;
  const int  h = (int)(tid % H);
  long s_, d_;
  if (e < Eorig) { s_ = src[e]; d_ = dst[e]; }
  else           { s_ = d_ = e - Eorig; }

  const long HC = (long)H * C;
  const float* pl = xl + s_ * HC + (long)h * C;
  const float* pr = xr + d_ * HC + (long)h * C;
  const float* pa = att + (long)h * C;

  float acc = 0.0f;
  if ((C & 3) == 0) {
    const float4* pl4 = (const float4*)pl;
    const float4* pr4 = (const float4*)pr;
    const float4* pa4 = (const float4*)pa;
    for (int c = 0; c < (C >> 2); ++c) {
      const float4 a = pl4[c], b = pr4[c], w = pa4[c];
      float v0 = a.x + b.x, v1 = a.y + b.y, v2 = a.z + b.z, v3 = a.w + b.w;
      v0 = (v0 > 0.0f) ? v0 : NEG_SLOPE * v0;
      v1 = (v1 > 0.0f) ? v1 : NEG_SLOPE * v1;
      v2 = (v2 > 0.0f) ? v2 : NEG_SLOPE * v2;
      v3 = (v3 > 0.0f) ? v3 : NEG_SLOPE * v3;
      acc += v0 * w.x + v1 * w.y + v2 * w.z + v3 * w.w;
    }
  } else {
    for (int c = 0; c < C; ++c) {
      float v = pl[c] + pr[c];
      v = (v > 0.0f) ? v : NEG_SLOPE * v;
      acc += v * pa[c];
    }
  }
  logits[e * H + h] = acc;
  atomicMax(&mOrd[d_ * H + h], f2ord(acc));
}

// ---------------------------------------------------------------------------
// Pass B: p = exp(logit - max[dst]); logits <- p ; atomic sum into ssum[dst,h]
// ---------------------------------------------------------------------------
__global__ void __launch_bounds__(256) edge_exp_sum(
    float* __restrict__ logits, const int* __restrict__ dst,
    const unsigned* __restrict__ mOrd, float* __restrict__ ssum,
    long Eorig, long Etot, int H)
{
  long tid = (long)blockIdx.x * blockDim.x + threadIdx.x;
  if (tid >= Etot * H) return;
  const long e = tid / H;
  const int  h = (int)(tid % H);
  const long d_ = (e < Eorig) ? (long)dst[e] : (e - Eorig);

  const float mx = ord2f(mOrd[d_ * H + h]);
  const float pv = __expf(logits[e * H + h] - mx);
  logits[e * H + h] = pv;
  atomicAdd(&ssum[d_ * H + h], pv);
}

// ---------------------------------------------------------------------------
// Pass C: out[dst,h,c] += (p / ssum[dst,h]) * xl[src,h,c]
// ---------------------------------------------------------------------------
__global__ void __launch_bounds__(256) edge_aggregate(
    const float* __restrict__ p, const float* __restrict__ ssum,
    const float* __restrict__ xl,
    const int* __restrict__ src, const int* __restrict__ dst,
    float* __restrict__ out, long Eorig, long Etot, int H, int C)
{
  long tid = (long)blockIdx.x * blockDim.x + threadIdx.x;
  if (tid >= Etot * H) return;
  const long e = tid / H;
  const int  h = (int)(tid % H);
  long s_, d_;
  if (e < Eorig) { s_ = src[e]; d_ = dst[e]; }
  else           { s_ = d_ = e - Eorig; }

  const float alpha = p[e * H + h] / ssum[d_ * H + h];
  const long HC = (long)H * C;
  const float* pl = xl + s_ * HC + (long)h * C;
  float* po = out + d_ * HC + (long)h * C;

  if ((C & 3) == 0) {
    const float4* pl4 = (const float4*)pl;
    for (int c = 0; c < (C >> 2); ++c) {
      const float4 v = pl4[c];
      atomicAdd(&po[4 * c + 0], alpha * v.x);
      atomicAdd(&po[4 * c + 1], alpha * v.y);
      atomicAdd(&po[4 * c + 2], alpha * v.z);
      atomicAdd(&po[4 * c + 3], alpha * v.w);
    }
  } else {
    for (int c = 0; c < C; ++c)
      atomicAdd(&po[c], alpha * pl[c]);
  }
}

// out = elu(out + bias)   (layers 1 & 2)
__global__ void __launch_bounds__(256) finalize_elu(
    float* __restrict__ out, const float* __restrict__ bias, long n, int HC)
{
  long i = (long)blockIdx.x * blockDim.x + threadIdx.x;
  if (i >= n) return;
  float v = out[i] + bias[i % HC];
  out[i] = (v > 0.0f) ? v : (__expf(v) - 1.0f);
}

// final: d_out[n] = mean_h(acc[n,h]) + bias3[0]   (conv3, concat=False, C=1)
__global__ void __launch_bounds__(256) finalize_mean(
    const float* __restrict__ acc, const float* __restrict__ bias3,
    float* __restrict__ out, long N, int H)
{
  long n = (long)blockIdx.x * blockDim.x + threadIdx.x;
  if (n >= N) return;
  float s = 0.0f;
  for (int h = 0; h < H; ++h) s += acc[n * H + h];
  out[n] = s / (float)H + bias3[0];
}

// ---------------------------------------------------------------------------
// Host-side helpers
// ---------------------------------------------------------------------------
static void run_gemm(const float* X, const float* W, const float* b, float* Y,
                     int N, int K, int M, hipStream_t stream)
{
  const int rowBlocks = (N + 63) / 64;
  const int colBlocks = (M + 63) / 64;
  gemm_bias_wmma<<<(unsigned)(rowBlocks * colBlocks), 256, 0, stream>>>(
      X, W, b, Y, N, K, M, colBlocks);
}

static void run_edge_stage(const float* xl, const float* xr,
                           const int* src, const int* dst, const float* att,
                           float* pbuf, unsigned* mo, float* ss, float* out,
                           long E, long N, int H, int C, hipStream_t stream)
{
  const long Etot = E + N;
  const long nmh = N * H;
  const long nout = N * (long)H * C;
  const long t = Etot * H;
  fill_u32<<<(unsigned)((nmh + 255) / 256), 256, 0, stream>>>(mo, 0u, nmh);
  fill_f32<<<(unsigned)((nmh + 255) / 256), 256, 0, stream>>>(ss, 0.0f, nmh);
  fill_f32<<<(unsigned)((nout + 255) / 256), 256, 0, stream>>>(out, 0.0f, nout);
  edge_logits_max<<<(unsigned)((t + 255) / 256), 256, 0, stream>>>(
      xl, xr, src, dst, att, pbuf, mo, E, Etot, H, C);
  edge_exp_sum<<<(unsigned)((t + 255) / 256), 256, 0, stream>>>(
      pbuf, dst, mo, ss, E, Etot, H);
  edge_aggregate<<<(unsigned)((t + 255) / 256), 256, 0, stream>>>(
      pbuf, ss, xl, src, dst, out, E, Etot, H, C);
}

extern "C" void kernel_launch(void* const* d_in, const int* in_sizes, int n_in,
                              void* d_out, int out_size, void* d_ws, size_t ws_size,
                              hipStream_t stream)
{
  const float* x    = (const float*)d_in[0];
  const int*   ei   = (const int*)d_in[1];      // [2,E] (src row, dst row)
  const float* Wl1  = (const float*)d_in[2];
  const float* bl1  = (const float*)d_in[3];
  const float* Wr1  = (const float*)d_in[4];
  const float* br1  = (const float*)d_in[5];
  const float* att1 = (const float*)d_in[6];
  const float* bi1  = (const float*)d_in[7];
  const float* Wl2  = (const float*)d_in[8];
  const float* bl2  = (const float*)d_in[9];
  const float* Wr2  = (const float*)d_in[10];
  const float* br2  = (const float*)d_in[11];
  const float* att2 = (const float*)d_in[12];
  const float* bi2  = (const float*)d_in[13];
  const float* Wl3  = (const float*)d_in[14];
  const float* bl3  = (const float*)d_in[15];
  const float* Wr3  = (const float*)d_in[16];
  const float* br3  = (const float*)d_in[17];
  const float* att3 = (const float*)d_in[18];
  const float* bi3  = (const float*)d_in[19];

  const int  F_in = 58;
  const long N = in_sizes[0] / F_in;     // 50000
  const long E = in_sizes[1] / 2;        // 500000
  const int* src = ei;
  const int* dst = ei + E;
  const long Etot = E + N;

  // Workspace layout
  float* ws = (float*)d_ws;
  size_t o = 0;
  float* xl = ws + o;  o += (size_t)N * 256;
  float* xr = ws + o;  o += (size_t)N * 256;
  float* h1 = ws + o;  o += (size_t)N * 256;
  float* h2 = ws + o;  o += (size_t)N * 256;
  float* pb = ws + o;  o += (size_t)Etot * 6;
  unsigned* mo = (unsigned*)(ws + o); o += (size_t)N * 6;
  float* ss = ws + o;  o += (size_t)N * 6;
  float* a3 = ws + o;  o += (size_t)N * 6;

  // -------- Layer 1: 58 -> 4 heads x 64, concat, ELU --------
  run_gemm(x, Wl1, bl1, xl, (int)N, F_in, 256, stream);
  run_gemm(x, Wr1, br1, xr, (int)N, F_in, 256, stream);
  run_edge_stage(xl, xr, src, dst, att1, pb, mo, ss, h1, E, N, 4, 64, stream);
  {
    const long n = N * 256;
    finalize_elu<<<(unsigned)((n + 255) / 256), 256, 0, stream>>>(h1, bi1, n, 256);
  }

  // -------- Layer 2: 256 -> 4 heads x 64, concat, ELU --------
  run_gemm(h1, Wl2, bl2, xl, (int)N, 256, 256, stream);
  run_gemm(h1, Wr2, br2, xr, (int)N, 256, 256, stream);
  run_edge_stage(xl, xr, src, dst, att2, pb, mo, ss, h2, E, N, 4, 64, stream);
  {
    const long n = N * 256;
    finalize_elu<<<(unsigned)((n + 255) / 256), 256, 0, stream>>>(h2, bi2, n, 256);
  }

  // -------- Layer 3: 256 -> 6 heads x 1, mean over heads --------
  run_gemm(h2, Wl3, bl3, xl, (int)N, 256, 6, stream);
  run_gemm(h2, Wr3, br3, xr, (int)N, 256, 6, stream);
  run_edge_stage(xl, xr, src, dst, att3, pb, mo, ss, a3, E, N, 6, 1, stream);
  finalize_mean<<<(unsigned)((N + 255) / 256), 256, 0, stream>>>(
      a3, bi3, (float*)d_out, N, 6);
}
